// LSTM_5308579578353
// MI455X (gfx1250) — matrix operand order repaired and checked by reading, not compiled, not run on hardware
//
#include <hip/hip_runtime.h>

#define BATCH  64
#define MAXLEN 512
#define EMB    1024
#define HID    1024
#define GATES  4096   // 4*HID
#define BT     (BATCH * MAXLEN)   // 32768

typedef __attribute__((ext_vector_type(16))) __bf16 bf16x16;
typedef __attribute__((ext_vector_type(8)))  float  f32x8;
typedef __attribute__((ext_vector_type(4)))  unsigned int u32x4;
typedef __attribute__((ext_vector_type(4)))  int i32x4;
typedef __attribute__((ext_vector_type(8)))  int i32x8;

union FragBF { bf16x16 v; int4 q[2]; };

#if defined(__has_builtin)
#if __has_builtin(__builtin_amdgcn_tensor_load_to_lds)
#define USE_TDM 1
#endif
#endif
#ifndef USE_TDM
#define USE_TDM 0
#endif

__device__ __forceinline__ unsigned short f32_to_bf16(float f) {
    unsigned int u = __float_as_uint(f);
    u += 0x7FFFu + ((u >> 16) & 1u);      // round-to-nearest-even
    return (unsigned short)(u >> 16);
}

__device__ __forceinline__ float sigm(float x)   { return 1.0f / (1.0f + __expf(-x)); }
__device__ __forceinline__ float tanh_f(float x) { return 2.0f / (1.0f + __expf(-2.0f * x)) - 1.0f; }

#if USE_TDM
// 2D bf16 tile (tile_d0 contiguous elems x tile_d1 rows), row stride `stride`
// elems; LDS rows padded by TDM: 64B data + 16B pad (PADK = 40 elems).
__device__ __forceinline__ void tdm_load_2d(unsigned lds_addr, const void* gptr,
                                            unsigned tile_d0, unsigned tile_d1,
                                            unsigned stride, unsigned tensor_d0)
{
    unsigned long long ga = (unsigned long long)gptr;
    u32x4 g0;
    g0.x = 1u;                                                // count=1, user mode
    g0.y = lds_addr;                                          // LDS byte address
    g0.z = (unsigned)(ga & 0xFFFFFFFFu);                      // global_addr[31:0]
    g0.w = (unsigned)((ga >> 32) & 0x01FFFFFFu) | (2u << 30); // addr[56:32] | type=2
    i32x8 g1;
    g1[0] = (1 << 16)      // data_size = 2 bytes
          | (1 << 20)      // pad_enable
          | (3 << 22)      // pad_interval: 16 dwords (64B) of data
          | (3 << 25);     // pad_amount:   4 dwords (16B) of pad
    g1[1] = (int)(tensor_d0 << 16);                           // tensor_dim0[15:0]
    g1[2] = (int)((tensor_d0 >> 16) | (0x100000u << 16));     // dim0 hi | tensor_dim1 lo
    g1[3] = (int)((0x100000u >> 16) | (tile_d0 << 16));       // tensor_dim1 hi | tile_dim0
    g1[4] = (int)(tile_d1 & 0xFFFFu);                         // tile_dim1 (tile_dim2=0)
    g1[5] = (int)stride;                                      // tensor_dim0_stride lo
    g1[6] = 0; g1[7] = 0;
    i32x4 z4 = {0, 0, 0, 0};
#if __clang_major__ >= 23
    i32x8 z8 = {0, 0, 0, 0, 0, 0, 0, 0};
    __builtin_amdgcn_tensor_load_to_lds(g0, g1, z4, z4, z8, 0);
#else
    __builtin_amdgcn_tensor_load_to_lds(g0, g1, z4, z4, 0);
#endif
}
#endif

// ---------------------------------------------------------------- converters
__global__ void cvt_bf16(const float* __restrict__ in, unsigned short* __restrict__ out, int n) {
    int i = blockIdx.x * blockDim.x + threadIdx.x;
    int stride = gridDim.x * blockDim.x;
    for (; i < n; i += stride) out[i] = f32_to_bf16(in[i]);
}

__global__ void zero_u32(unsigned int* __restrict__ p, int n) {
    int i = blockIdx.x * blockDim.x + threadIdx.x;
    if (i < n) p[i] = 0u;
}

// ---------------------------------------------------------------- big GEMM
// C[m][n] = sum_k A[m][k]*B[n][k]  (A,B bf16 row-major along K).
// Workgroup tile 128(M) x 256(N), K-step 32; 8 waves as 2(M) x 4(N),
// each wave a square 64x64 tile: 8 fragments per 16 WMMAs.
// Tiles staged by the Tensor Data Mover, double-buffered on TENSORcnt.
#define TM 128
#define TN 256
#define TK 32
#define PADK 40   // 32 elems + 8 elem pad (TDM pad: 64B data + 16B pad)

__global__ __launch_bounds__(256) void gemm_bf16_nt(
    const unsigned short* __restrict__ A,
    const unsigned short* __restrict__ B,
    float* __restrict__ C,
    int M, int N, int K)
{
    __shared__ unsigned short sA[2][TM * PADK];   // 20 KB
    __shared__ unsigned short sB[2][TN * PADK];   // 40 KB

    const int m0 = blockIdx.y * TM;
    const int n0 = blockIdx.x * TN;
    const int tid = threadIdx.x;
    const int lane = tid & 31;
    const int w  = tid >> 5;
    const int wm = w & 1;      // 2 waves along M (64 rows each)
    const int wn = w >> 1;     // 4 waves along N (64 cols each)
    const int nk = K / TK;

    f32x8 acc[4][4];
#pragma unroll
    for (int i = 0; i < 4; ++i)
#pragma unroll
        for (int j = 0; j < 4; ++j)
#pragma unroll
            for (int r = 0; r < 8; ++r) acc[i][j][r] = 0.0f;

#if USE_TDM
    if (w == 0) {   // TDM ignores EXEC; one issue per workgroup
        tdm_load_2d((unsigned)(unsigned long long)&sA[0][0], A + m0 * K, TK, TM, K, K);
        tdm_load_2d((unsigned)(unsigned long long)&sB[0][0], B + n0 * K, TK, TN, K, K);
    }
#endif

    const int arow = tid >> 2;        // 0..63
    const int acol = (tid & 3) * 8;   // 0,8,16,24

    for (int i = 0; i < nk; ++i) {
        const int buf = i & 1;
#if USE_TDM
        if (w == 0) {
            if (i + 1 < nk) {
                const int k1 = (i + 1) * TK;
                tdm_load_2d((unsigned)(unsigned long long)&sA[buf ^ 1][0], A + m0 * K + k1, TK, TM, K, K);
                tdm_load_2d((unsigned)(unsigned long long)&sB[buf ^ 1][0], B + n0 * K + k1, TK, TN, K, K);
                __builtin_amdgcn_s_wait_tensorcnt(2);   // current tile pair done
            } else {
                __builtin_amdgcn_s_wait_tensorcnt(0);
            }
        }
        __syncthreads();
#else
        { // cooperative fallback staging
            const int k0 = i * TK;
            *(int4*)&sA[buf][arow * PADK + acol]        = *(const int4*)&A[(m0 + arow) * K + k0 + acol];
            *(int4*)&sA[buf][(arow + 64) * PADK + acol] = *(const int4*)&A[(m0 + arow + 64) * K + k0 + acol];
#pragma unroll
            for (int r = 0; r < 4; ++r) {
                int brow = arow + 64 * r;
                *(int4*)&sB[buf][brow * PADK + acol] = *(const int4*)&B[(n0 + brow) * K + k0 + acol];
            }
            __syncthreads();
        }
#endif
        const unsigned short* a_ = sA[buf];
        const unsigned short* b_ = sB[buf];

        FragBF afr[4], bfr[4];
#pragma unroll
        for (int ms = 0; ms < 4; ++ms) {
            int row = wm * 64 + ms * 16 + (lane & 15);
            int kb  = (lane >> 4) * 8;
            afr[ms].q[0] = *(const int4*)&a_[row * PADK + kb];
            afr[ms].q[1] = *(const int4*)&a_[row * PADK + kb + 16];
        }
#pragma unroll
        for (int ns = 0; ns < 4; ++ns) {
            int row = wn * 64 + ns * 16 + (lane & 15);
            int kb  = (lane >> 4) * 8;
            bfr[ns].q[0] = *(const int4*)&b_[row * PADK + kb];
            bfr[ns].q[1] = *(const int4*)&b_[row * PADK + kb + 16];
        }
#pragma unroll
        for (int ns = 0; ns < 4; ++ns)
#pragma unroll
            for (int ms = 0; ms < 4; ++ms)
                acc[ms][ns] = __builtin_amdgcn_wmma_f32_16x16x32_bf16(
                    false, afr[ms].v, false, bfr[ns].v, (short)0, acc[ms][ns], false, false);
        __syncthreads();
    }

    // C/D layout: lane<16 -> N=lane, VGPR r: M=r ; lane>=16 -> N=lane-16, M=8+r
#pragma unroll
    for (int ms = 0; ms < 4; ++ms)
#pragma unroll
        for (int ns = 0; ns < 4; ++ns) {
            int n  = n0 + wn * 64 + ns * 16 + (lane & 15);
            int mb = m0 + wm * 64 + ms * 16 + (lane >> 4) * 8;
#pragma unroll
            for (int r = 0; r < 8; ++r)
                C[(mb + r) * N + n] = acc[ms][ns][r];
        }
}

// ---------------------------------------------------------------- recurrent step
// HTILE=16 hidden units per workgroup. Waves 0..2 compute gh for gates {i,g,o}
// (h_f is unused by the reference). M=64, N=16, K=1024. The shared h-tile is
// TDM-staged into LDS once per K-step (reused by all 3 waves); W_hh fragments
// come straight from the L2-resident weight copy.
#define HTILE 16
#define NKH   (HID / 32)

__global__ __launch_bounds__(128) void lstm_step(
    const unsigned short* __restrict__ h_in_bf16,   // 64 x 1024
    unsigned short* __restrict__ h_out_bf16,        // 64 x 1024 (ping-pong)
    const unsigned short* __restrict__ Whh,         // 4096 x 1024 bf16
    const float* __restrict__ gi_all,               // (B*T) x 4096
    const int* __restrict__ lengths,                // 64
    float* __restrict__ h_state,                    // 64 x 1024 f32
    float* __restrict__ c_state,                    // 64 x 1024 f32
    float* __restrict__ f_seq,                      // B x T x H (d_out)
    int t)
{
    __shared__ float gh_s[3][64][20];               // padded, 15.4 KB
#if USE_TDM
    __shared__ unsigned short sH[2][64 * PADK];     // 10 KB, TDM-padded h tiles
#endif

    const int h0   = blockIdx.x * HTILE;
    const int tid  = threadIdx.x;
    const int w    = tid >> 5;
    const int lane = tid & 31;

    const int gate = (w == 0) ? 0 : (w == 1 ? 2 : 3);   // i, g, o (w==3 unused)
    const int bn   = gate * HID + h0 + (lane & 15);
    const int kA0  = (lane >> 4) * 8;
    const int kB0  = (lane >> 4) * 8;

    f32x8 acc[4];
#pragma unroll
    for (int ms = 0; ms < 4; ++ms)
#pragma unroll
        for (int r = 0; r < 8; ++r) acc[ms][r] = 0.0f;

#if USE_TDM
    if (w == 0)
        tdm_load_2d((unsigned)(unsigned long long)&sH[0][0], h_in_bf16, 32, 64, HID, HID);
#endif

    for (int i = 0; i < NKH; ++i) {
        const int k0 = i * 32;
#if USE_TDM
        const int buf = i & 1;
        if (w == 0) {
            if (i + 1 < NKH) {
                tdm_load_2d((unsigned)(unsigned long long)&sH[buf ^ 1][0],
                            h_in_bf16 + (i + 1) * 32, 32, 64, HID, HID);
                __builtin_amdgcn_s_wait_tensorcnt(1);
            } else {
                __builtin_amdgcn_s_wait_tensorcnt(0);
            }
        }
        __syncthreads();
#endif
        if (w < 3) {
            __builtin_prefetch(&Whh[bn * HID + ((k0 + 32) & (HID - 1)) + kB0], 0, 1);
            FragBF bfr, afr[4];
            bfr.q[0] = *(const int4*)&Whh[bn * HID + k0 + kB0];
            bfr.q[1] = *(const int4*)&Whh[bn * HID + k0 + kB0 + 16];
#pragma unroll
            for (int ms = 0; ms < 4; ++ms) {
                int m = ms * 16 + (lane & 15);
#if USE_TDM
                afr[ms].q[0] = *(const int4*)&sH[buf][m * PADK + kA0];
                afr[ms].q[1] = *(const int4*)&sH[buf][m * PADK + kA0 + 16];
#else
                afr[ms].q[0] = *(const int4*)&h_in_bf16[m * HID + k0 + kA0];
                afr[ms].q[1] = *(const int4*)&h_in_bf16[m * HID + k0 + kA0 + 16];
#endif
            }
#pragma unroll
            for (int ms = 0; ms < 4; ++ms)
                acc[ms] = __builtin_amdgcn_wmma_f32_16x16x32_bf16(
                    false, afr[ms].v, false, bfr.v, (short)0, acc[ms], false, false);
        }
#if USE_TDM
        __syncthreads();
#endif
    }

    if (w < 3) {
#pragma unroll
        for (int ms = 0; ms < 4; ++ms) {
            int mb = ms * 16 + (lane >> 4) * 8;
            int n  = lane & 15;
#pragma unroll
            for (int r = 0; r < 8; ++r) gh_s[w][mb + r][n] = acc[ms][r];
        }
    }
    __syncthreads();

    for (int e = tid; e < 64 * HTILE; e += 128) {
        int b   = e >> 4;
        int j   = e & 15;
        int hid = h0 + j;
        const float* gi = gi_all + (b * MAXLEN + t) * GATES;

        float i_i = gi[hid];
        float i_f = gi[HID + hid];
        float i_g = gi[2 * HID + hid];
        float i_o = gi[3 * HID + hid];
        float h_i = gh_s[0][b][j];
        float h_g = gh_s[1][b][j];
        float h_o = gh_s[2][b][j];
        float c   = c_state[b * HID + hid];
        float h   = h_state[b * HID + hid];

        float gx_i = sigm(i_i);
        float fx_i = gx_i * (1.0f - gx_i);
        float gx_f = sigm(i_f);
        float gx_o = sigm(i_o);
        float fx_o = gx_o * (1.0f - gx_o);
        float gx_c = tanh_f(i_g);
        float fx_c = 1.0f - gx_c * gx_c;
        float g_c  = gx_i * gx_c;
        float g_ct = tanh_f(g_c);
        float g_h  = gx_o * g_ct;
        float Bt   = gx_f * c;
        float Dt   = gx_c * fx_i * h_i + gx_i * fx_c * h_g;
        float s2   = 1.0f - g_ct * g_ct;
        float Et   = gx_o * s2 * gx_f * c;
        float Ft   = gx_o * s2 * Dt + fx_o * g_ct * h_o;
        float h_new = Et + Ft + ((t == 0) ? g_h : 0.0f);
        float c_new = Bt + Dt + ((t == 0) ? g_c : 0.0f);

        bool active = (t < lengths[b]);
        float h2 = active ? h_new : h;
        float c2 = active ? c_new : c;

        c_state[b * HID + hid] = c2;
        h_state[b * HID + hid] = h2;
        h_out_bf16[b * HID + hid] = f32_to_bf16(h2);
        f_seq[(b * MAXLEN + t) * HID + hid] = h2;
    }
}

__global__ void copy_hfinal(const float* __restrict__ h_state, float* __restrict__ out, int n) {
    int i = blockIdx.x * blockDim.x + threadIdx.x;
    if (i < n) out[i] = h_state[i];
}

// ---------------------------------------------------------------- launcher
extern "C" void kernel_launch(void* const* d_in, const int* in_sizes, int n_in,
                              void* d_out, int out_size, void* d_ws, size_t ws_size,
                              hipStream_t stream)
{
    const float* x      = (const float*)d_in[0];
    const int*  lengths = (const int*)d_in[1];
    const float* W_ih   = (const float*)d_in[2];
    const float* W_hh   = (const float*)d_in[3];
    float* out = (float*)d_out;

    char* ws = (char*)d_ws;
    size_t off = 0;
    float* gi_all = (float*)(ws + off);                   off += (size_t)BT * GATES * 4;     // 512 MB
    unsigned short* x_bf   = (unsigned short*)(ws + off); off += (size_t)BT * EMB * 2;       // 64 MB
    unsigned short* wih_bf = (unsigned short*)(ws + off); off += (size_t)GATES * EMB * 2;    // 8 MB
    unsigned short* whh_bf = (unsigned short*)(ws + off); off += (size_t)GATES * HID * 2;    // 8 MB
    float* h_state = (float*)(ws + off);                  off += (size_t)BATCH * HID * 4;
    float* c_state = (float*)(ws + off);                  off += (size_t)BATCH * HID * 4;
    unsigned short* hbf0 = (unsigned short*)(ws + off);   off += (size_t)BATCH * HID * 2;
    unsigned short* hbf1 = (unsigned short*)(ws + off);   off += (size_t)BATCH * HID * 2;

    cvt_bf16<<<2048, 256, 0, stream>>>(x, x_bf, BT * EMB);
    cvt_bf16<<<1024, 256, 0, stream>>>(W_ih, wih_bf, GATES * EMB);
    cvt_bf16<<<1024, 256, 0, stream>>>(W_hh, whh_bf, GATES * HID);

    zero_u32<<<768, 256, 0, stream>>>((unsigned int*)h_state, 196608);

    gemm_bf16_nt<<<dim3(GATES / TN, BT / TM), 256, 0, stream>>>(x_bf, wih_bf, gi_all, BT, GATES, EMB);

    for (int t = 0; t < MAXLEN; ++t) {
        const unsigned short* hin = (t & 1) ? hbf1 : hbf0;
        unsigned short*      hout = (t & 1) ? hbf0 : hbf1;
        lstm_step<<<HID / HTILE, 128, 0, stream>>>(hin, hout, whh_bf, gi_all, lengths,
                                                   h_state, c_state, out, t);
    }

    copy_hfinal<<<256, 256, 0, stream>>>(h_state, out + (size_t)BT * HID, BATCH * HID);
}
